// MultiScaleFreqRWKV_90580860273236
// MI455X (gfx1250) — matrix operand order repaired and checked by reading, not compiled, not run on hardware
//
#include <hip/hip_runtime.h>
#include <hip/hip_bf16.h>
#include <stdint.h>

typedef __bf16 bf16_t;
typedef __attribute__((ext_vector_type(4)))  __bf16 v4bf;
typedef __attribute__((ext_vector_type(8)))  __bf16 v8bf;
typedef __attribute__((ext_vector_type(16))) __bf16 v16bf;
typedef __attribute__((ext_vector_type(8)))  float  v8f;
typedef unsigned int u32x4 __attribute__((ext_vector_type(4)));
typedef int          i32x4 __attribute__((ext_vector_type(4)));
typedef int          i32x8 __attribute__((ext_vector_type(8)));

#if __has_builtin(__builtin_amdgcn_tensor_load_to_lds) && __has_builtin(__builtin_amdgcn_s_wait_tensorcnt)
#define HAVE_TDM 1
#pragma message("CDNA5 probe: tensor_load_to_lds + s_wait_tensorcnt AVAILABLE -> TDM path enabled")
#else
#define HAVE_TDM 0
#pragma message("CDNA5 probe: tensor_load_to_lds NOT available -> b128 fallback path")
#endif

#if __has_builtin(__builtin_amdgcn_global_load_async_to_lds_b128)
#pragma message("CDNA5 probe: __builtin_amdgcn_global_load_async_to_lds_b128 AVAILABLE")
#else
#pragma message("CDNA5 probe: __builtin_amdgcn_global_load_async_to_lds_b128 NOT available")
#endif
#if __has_builtin(__builtin_amdgcn_cluster_load_async_to_lds_b128)
#pragma message("CDNA5 probe: __builtin_amdgcn_cluster_load_async_to_lds_b128 AVAILABLE")
#endif

// ---------------------------------------------------------------------------
// Weight packing: f32 [K,N] -> bf16 TRANSPOSED [N,K], so a 64-column GEMM
// panel is one contiguous slab (TDM/b128-friendly, no LDS scatter).
// ---------------------------------------------------------------------------
__global__ void pack1t_kernel(const float* __restrict__ src, bf16_t* __restrict__ dst,
                              int K, int N) {
  int i = blockIdx.x * 256 + threadIdx.x;      // over N*K, dst-linear
  if (i >= N * K) return;
  int n = i / K, k = i % K;
  dst[i] = (bf16_t)src[k * N + n];
}
// dst [384][128] = transposed stack of {Wk|Wv|Wr} (each [128,128])
__global__ void pack3t_kernel(const float* __restrict__ s0, const float* __restrict__ s1,
                              const float* __restrict__ s2, bf16_t* __restrict__ dst) {
  int i = blockIdx.x * 256 + threadIdx.x;      // over 384*128
  if (i >= 384 * 128) return;
  int n = i >> 7, k = i & 127;
  const float* s = (n < 128) ? s0 : (n < 256) ? s1 : s2;
  dst[i] = (bf16_t)s[k * 128 + (n & 127)];
}

// ---------------------------------------------------------------------------
// Time pooling for medium / coarse branches.
// ---------------------------------------------------------------------------
__global__ void pool2_kernel(const float* __restrict__ x, float* __restrict__ y) {
  int i = blockIdx.x * 256 + threadIdx.x;      // 16*2048*128
  int d = i & 127, t = (i >> 7) & 2047, b = i >> 18;
  size_t base = ((size_t)b * 4096 + (size_t)t * 2) * 256 + d;
  y[i] = 0.5f * (x[base] + x[base + 256]);
}
__global__ void pool4_kernel(const float* __restrict__ x, float* __restrict__ y) {
  int i = blockIdx.x * 256 + threadIdx.x;      // 16*1024*64
  int d = i & 63, t = (i >> 6) & 1023, b = i >> 16;
  size_t base = ((size_t)b * 4096 + (size_t)t * 4) * 256 + d;
  y[i] = 0.25f * (x[base] + x[base + 256] + x[base + 512] + x[base + 768]);
}

// ---------------------------------------------------------------------------
// LayerNorm over 128 channels, one wave32 per row (4 ch / lane).
// ---------------------------------------------------------------------------
__device__ inline void wave_ln_row(const float* __restrict__ x, const float* __restrict__ g,
                                   const float* __restrict__ b, float* __restrict__ out) {
  int lane = threadIdx.x & 31;
  int c0 = lane * 4;
  float4 xv = *(const float4*)(x + c0);
  float s  = xv.x + xv.y + xv.z + xv.w;
  float s2 = xv.x * xv.x + xv.y * xv.y + xv.z * xv.z + xv.w * xv.w;
  for (int off = 16; off; off >>= 1) { s += __shfl_xor(s, off); s2 += __shfl_xor(s2, off); }
  float mean = s * (1.f / 128.f);
  float var  = s2 * (1.f / 128.f) - mean * mean;
  float inv  = rsqrtf(var + 1e-5f);
  float4 gv = *(const float4*)(g + c0);
  float4 bv = *(const float4*)(b + c0);
  float4 o;
  o.x = (xv.x - mean) * inv * gv.x + bv.x;
  o.y = (xv.y - mean) * inv * gv.y + bv.y;
  o.z = (xv.z - mean) * inv * gv.z + bv.z;
  o.w = (xv.w - mean) * inv * gv.w + bv.w;
  *(float4*)(out + c0) = o;
}

__global__ __launch_bounds__(256) void ln_kernel(const float* __restrict__ h,
                                                 const float* __restrict__ g,
                                                 const float* __restrict__ b,
                                                 float* __restrict__ z) {
  int row = blockIdx.x * 8 + (threadIdx.x >> 5);
  wave_ln_row(h + (size_t)row * 128, g, b, z + (size_t)row * 128);
}

__global__ __launch_bounds__(512) void lastln_kernel(const float* __restrict__ h, int T,
                                                     const float* __restrict__ g,
                                                     const float* __restrict__ b,
                                                     float* __restrict__ out) {
  int bb = threadIdx.x >> 5;
  size_t row = (size_t)bb * T + (T - 1);
  wave_ln_row(h + row * 128, g, b, out + (size_t)bb * 128);
}

// ---------------------------------------------------------------------------
// WKV scan: sequential over T, one thread per (batch, channel). Emits
// ry = sigmoid(r) * y as bf16 (A-matrix of the Wo WMMA GEMM).
// ---------------------------------------------------------------------------
__global__ __launch_bounds__(256) void wkv_kernel(const float* __restrict__ kvr,
                                                  const float* __restrict__ decay,
                                                  const float* __restrict__ first,
                                                  bf16_t* __restrict__ ry, int T) {
  int gid = blockIdx.x * 256 + threadIdx.x;   // 0 .. 16*128
  int b = gid >> 7, ch = gid & 127;
  float w = __expf(decay[ch]);
  float u = first[ch];
  float aa = 0.f, bb = 0.f, pp = -1e38f;
  const float* base = kvr + (size_t)b * T * 384;
  bf16_t* rb = ry + (size_t)b * T * 128;
  for (int t = 0; t < T; ++t) {
    const float* rp = base + (size_t)t * 384;
    __builtin_prefetch(rp + 384 * 8, 0, 0);
    float kt = rp[ch], vt = rp[128 + ch], rt = rp[256 + ch];
    float q  = fmaxf(pp, u + kt);
    float e1 = __expf(pp - q), e2 = __expf(u + kt - q);
    float yy = (e1 * aa + e2 * vt) / (e1 * bb + e2);
    float q2 = fmaxf(pp - w, kt);
    e1 = __expf(pp - w - q2); e2 = __expf(kt - q2);
    aa = e1 * aa + e2 * vt; bb = e1 * bb + e2; pp = q2;
    float sr = 1.f / (1.f + __expf(-rt));
    rb[(size_t)t * 128 + ch] = (bf16_t)(sr * yy);
  }
}

// ---------------------------------------------------------------------------
// WMMA GEMM: C[M,N] = A[M,K] @ B[K,N], bf16 x bf16 -> f32 accumulate.
//  - B pre-packed bf16 transposed [N][K]; each block's 64-col panel is a
//    contiguous slab loaded to LDS ONCE via TENSOR_LOAD_TO_LDS (TDM) when
//    available (s_wait_tensorcnt), else b128 copies.
//  - A staged per 32-wide K chunk, double-buffered; f32->bf16 conversion
//    (v_cvt_pk_bf16_f32 pairs) + optional token-shift mix fused in.
//  - Block tile 128x64, 8 waves, wave tile 32x32 = 2x2 v_wmma_f32_16x16x32_bf16.
// ---------------------------------------------------------------------------
enum { AM_F32 = 0, AM_F32MIX = 1, AM_BF16 = 2 };
enum { EP_NONE = 0, EP_BIAS = 1, EP_SIGMOID = 2, EP_RELU2BF = 3, EP_ADD = 4, EP_MULADD = 5 };

template <int AM, int EP, int K>
__global__ __launch_bounds__(256) void gemm_kernel(
    const void* __restrict__ Av, const bf16_t* __restrict__ Bw,
    int M, int N, int Tmask,
    const float* __restrict__ mix0, const float* __restrict__ mix1, const float* __restrict__ mix2,
    const float* __restrict__ bias, const float* __restrict__ rr,
    float* __restrict__ outf, bf16_t* __restrict__ outb, float* __restrict__ hacc, int ldo) {
  __shared__ bf16_t sB[64 * K];        // whole B panel [64 cols][K]
  __shared__ bf16_t sA[2][128 * 32];   // double-buffered A chunk [row][k]
  const int tid  = threadIdx.x;
  const int bm   = blockIdx.x, bn = blockIdx.y;
  const int lane = tid & 31, wav = tid >> 5;
  const int wm = wav >> 1, wn = wav & 1;
  const int lrow = lane & 15, lhalf = lane >> 4;

  const float* mix = nullptr;
  if (AM == AM_F32MIX) {
    int msel = bn >> 1; if (msel > 2) msel = 2;    // 64-col blocks -> 128-col groups
    mix = (msel == 0) ? mix0 : (msel == 1) ? mix1 : mix2;
  }

  // -------- load entire B panel to LDS --------
  const bf16_t* bpanel = Bw + (size_t)bn * 64 * K;
#if HAVE_TDM
  if (wav == 0) {
    const unsigned int n = 64u * (unsigned int)K;          // elements (2B each), <= 32768
    uint64_t ga = (uint64_t)(uintptr_t)bpanel;
    unsigned int lb = (unsigned int)(uintptr_t)(&sB[0]);
    u32x4 g0; i32x8 g1; i32x4 gz;
    gz[0] = 0; gz[1] = 0; gz[2] = 0; gz[3] = 0;
    g0[0] = 1u;                                   // count=1, user descriptor
    g0[1] = lb;                                   // lds_addr
    g0[2] = (unsigned int)ga;                     // global_addr[31:0]
    g0[3] = (unsigned int)(ga >> 32) | (2u << 30);// global_addr[56:32] | type=2
    g1[0] = (int)(1u << 16);                      // data_size=1 (2 bytes)
    g1[1] = (int)(n << 16);                       // tensor_dim0[15:0]
    g1[2] = (int)(1u << 16);                      // tensor_dim0 hi=0 | tensor_dim1=1
    g1[3] = (int)(n << 16);                       // tile_dim0 = n (1-D tile)
    g1[4] = 0;                                    // tile_dim1/2 = 0
    g1[5] = (int)n;                               // tensor_dim0_stride[31:0]
    g1[6] = 0; g1[7] = 0;
#if __clang_major__ >= 23
    i32x8 g4; g4[0]=0; g4[1]=0; g4[2]=0; g4[3]=0; g4[4]=0; g4[5]=0; g4[6]=0; g4[7]=0;
    __builtin_amdgcn_tensor_load_to_lds(g0, g1, gz, gz, g4, 0);
#else
    __builtin_amdgcn_tensor_load_to_lds(g0, g1, gz, gz, 0);
#endif
    __builtin_amdgcn_s_wait_tensorcnt(0);
  }
#else
#pragma unroll
  for (int it = 0; it < (64 * K) / 2048; ++it) {
    int idx = it * 2048 + tid * 8;
    *(uint4*)&sB[idx] = *(const uint4*)(bpanel + idx);
  }
#endif

  // -------- A chunk staging (f32[+mix]->bf16 or bf16 copy) --------
  auto stageA = [&](int kc, int buf) {
    if (AM == AM_BF16) {
      const bf16_t* Ab = (const bf16_t*)Av;
      int r = tid >> 2, c8 = (tid & 3) * 8;
#pragma unroll
      for (int i = 0; i < 2; ++i) {
        int rl = r + i * 64;
        size_t grow = (size_t)bm * 128 + rl;
        uint4 vld = *(const uint4*)(Ab + grow * K + kc * 32 + c8);
        *(uint4*)&sA[buf][rl * 32 + c8] = vld;
      }
    } else {
      const float* Af = (const float*)Av;
      int r = tid >> 3, c4 = (tid & 7) * 4;
      int gk = kc * 32 + c4;
#pragma unroll
      for (int i = 0; i < 4; ++i) {
        int rl = r + i * 32;
        size_t grow = (size_t)bm * 128 + rl;
        float4 a = *(const float4*)(Af + grow * K + gk);
        if (AM == AM_F32MIX) {
          int t = (int)(grow & (size_t)Tmask);
          float4 p = make_float4(0.f, 0.f, 0.f, 0.f);
          if (t > 0) p = *(const float4*)(Af + (grow - 1) * K + gk);
          float4 mx = *(const float4*)(mix + gk);
          a.x = a.x * mx.x + p.x * (1.f - mx.x);
          a.y = a.y * mx.y + p.y * (1.f - mx.y);
          a.z = a.z * mx.z + p.z * (1.f - mx.z);
          a.w = a.w * mx.w + p.w * (1.f - mx.w);
        }
        // native bf16 vector -> v_cvt_pk_bf16_f32 x2 + single ds_store_b64
        v4bf cv = { (bf16_t)a.x, (bf16_t)a.y, (bf16_t)a.z, (bf16_t)a.w };
        *(v4bf*)&sA[buf][rl * 32 + c4] = cv;
      }
    }
  };

  const v8f vzero = {0.f, 0.f, 0.f, 0.f, 0.f, 0.f, 0.f, 0.f};
  v8f acc[2][2];
  acc[0][0] = vzero; acc[0][1] = vzero; acc[1][0] = vzero; acc[1][1] = vzero;

  constexpr int NK = K >> 5;
  stageA(0, 0);
  __syncthreads();                                // covers B panel + first A chunk
#pragma unroll
  for (int kc = 0; kc < NK; ++kc) {
    const int cur = kc & 1;
    v16bf af[2], bfr[2];
#pragma unroll
    for (int i = 0; i < 2; ++i) {
      int ar = wm * 32 + i * 16 + lrow;
      v8bf lo = *(const v8bf*)&sA[cur][ar * 32 + lhalf * 8];       // K = half*8..+7
      v8bf hi = *(const v8bf*)&sA[cur][ar * 32 + 16 + lhalf * 8];  // K = 16+half*8..+7
      af[i] = __builtin_shufflevector(lo, hi, 0, 1, 2, 3, 4, 5, 6, 7, 8, 9, 10, 11, 12, 13, 14, 15);
    }
#pragma unroll
    for (int j = 0; j < 2; ++j) {
      int bc = wn * 32 + j * 16 + lrow;
      v8bf lo = *(const v8bf*)&sB[bc * K + kc * 32 + lhalf * 8];       // K = half*8..+7
      v8bf hi = *(const v8bf*)&sB[bc * K + kc * 32 + 16 + lhalf * 8];  // K = 16+half*8..+7
      bfr[j] = __builtin_shufflevector(lo, hi, 0, 1, 2, 3, 4, 5, 6, 7, 8, 9, 10, 11, 12, 13, 14, 15);
    }
    // stage next chunk into the other buffer while wmma runs
    if (kc + 1 < NK) stageA(kc + 1, cur ^ 1);
#pragma unroll
    for (int i = 0; i < 2; ++i)
#pragma unroll
      for (int j = 0; j < 2; ++j)
        acc[i][j] = __builtin_amdgcn_wmma_f32_16x16x32_bf16(
            false, af[i], false, bfr[j], (short)0, acc[i][j], false, false);
    __syncthreads();
  }

  // -------- epilogue --------
#pragma unroll
  for (int i = 0; i < 2; ++i) {
#pragma unroll
    for (int j = 0; j < 2; ++j) {
#pragma unroll
      for (int r = 0; r < 8; ++r) {
        size_t grow = (size_t)bm * 128 + wm * 32 + i * 16 + lhalf * 8 + r;
        int gcol = bn * 64 + wn * 32 + j * 16 + lrow;
        float v = acc[i][j][r];
        if (EP == EP_NONE) {
          outf[grow * ldo + gcol] = v;
        } else if (EP == EP_BIAS) {
          outf[grow * ldo + gcol] = v + bias[gcol];
        } else if (EP == EP_SIGMOID) {
          outf[grow * ldo + gcol] = 1.f / (1.f + __expf(-v));
        } else if (EP == EP_RELU2BF) {
          float m = fmaxf(v, 0.f);
          outb[grow * ldo + gcol] = (bf16_t)(m * m);
        } else if (EP == EP_ADD) {
          hacc[grow * 128 + gcol] += v;
        } else if (EP == EP_MULADD) {
          hacc[grow * 128 + gcol] += rr[grow * 128 + gcol] * v;
        }
      }
    }
  }
}

// ---------------------------------------------------------------------------
// Readout: concat last tokens -> proj -> (softmax over len-1 == 1, q/k dead)
// att = (proj@Wv.T+bv)@Wo.T+bo -> LN. One block per batch row.
// ---------------------------------------------------------------------------
__global__ __launch_bounds__(512) void readout_kernel(
    const float* __restrict__ last, const float* __restrict__ Wp, const float* __restrict__ bp,
    const float* __restrict__ Wv, const float* __restrict__ bv,
    const float* __restrict__ Wo, const float* __restrict__ bo,
    const float* __restrict__ g, const float* __restrict__ beta, float* __restrict__ out) {
  __shared__ float comb[384], proj[512], vv[512], att[512], partial[32];
  int b = blockIdx.x, j = threadIdx.x;
  if (j < 384) { int grp = j >> 7, c = j & 127; comb[j] = last[(size_t)grp * 16 * 128 + b * 128 + c]; }
  __syncthreads();
  float acc = bp[j];
  for (int i = 0; i < 384; ++i) acc += comb[i] * Wp[(size_t)i * 512 + j];
  proj[j] = acc; __syncthreads();
  acc = bv[j];
  for (int i = 0; i < 512; ++i) acc += proj[i] * Wv[(size_t)j * 512 + i];
  vv[j] = acc; __syncthreads();
  acc = bo[j];
  for (int i = 0; i < 512; ++i) acc += vv[i] * Wo[(size_t)j * 512 + i];
  att[j] = acc; __syncthreads();
  float s = acc, s2 = acc * acc;
  for (int off = 16; off; off >>= 1) { s += __shfl_xor(s, off); s2 += __shfl_xor(s2, off); }
  int wav = j >> 5, lane = j & 31;
  if (lane == 0) { partial[wav] = s; partial[16 + wav] = s2; }
  __syncthreads();
  if (j == 0) {
    float ts = 0.f, ts2 = 0.f;
    for (int i = 0; i < 16; ++i) { ts += partial[i]; ts2 += partial[16 + i]; }
    partial[0] = ts; partial[16] = ts2;
  }
  __syncthreads();
  float mean = partial[0] * (1.f / 512.f);
  float var  = partial[16] * (1.f / 512.f) - mean * mean;
  out[(size_t)b * 512 + j] = (att[j] - mean) * rsqrtf(var + 1e-5f) * g[j] + beta[j];
}

// ---------------------------------------------------------------------------
// Host side
// ---------------------------------------------------------------------------
struct RwkvP {
  const float *w_in, *b_in, *ln1_g, *ln1_b, *ln2_g, *ln2_b, *mix_k, *mix_v, *mix_r,
              *decay, *first, *Wk, *Wv, *Wr, *Wo, *cmix_k, *cmix_r, *cWk, *cWr, *cWv,
              *lnf_g, *lnf_b;
};
static RwkvP unpack_rwkv(void* const* d, int& i) {
  RwkvP p;
  p.w_in  = (const float*)d[i++]; p.b_in  = (const float*)d[i++];
  p.ln1_g = (const float*)d[i++]; p.ln1_b = (const float*)d[i++];
  p.ln2_g = (const float*)d[i++]; p.ln2_b = (const float*)d[i++];
  p.mix_k = (const float*)d[i++]; p.mix_v = (const float*)d[i++]; p.mix_r = (const float*)d[i++];
  p.decay = (const float*)d[i++]; p.first = (const float*)d[i++];
  p.Wk = (const float*)d[i++]; p.Wv = (const float*)d[i++];
  p.Wr = (const float*)d[i++]; p.Wo = (const float*)d[i++];
  p.cmix_k = (const float*)d[i++]; p.cmix_r = (const float*)d[i++];
  p.cWk = (const float*)d[i++]; p.cWr = (const float*)d[i++]; p.cWv = (const float*)d[i++];
  p.lnf_g = (const float*)d[i++]; p.lnf_b = (const float*)d[i++];
  return p;
}
static inline int cdiv(int a, int b) { return (a + b - 1) / b; }

template <int DIN, int T>
static void run_branch(const float* x, const RwkvP& P,
                       float* h, float* z, float* kvr, bf16_t* ry,
                       bf16_t* kk, float* rrb, bf16_t* wbuf,
                       float* last_out, hipStream_t s) {
  constexpr int M = 16 * T;
  constexpr int Tm = T - 1;   // power-of-two mask: t = row & Tm
  // h = x @ w_in + b_in
  pack1t_kernel<<<cdiv(DIN * 128, 256), 256, 0, s>>>(P.w_in, wbuf, DIN, 128);
  gemm_kernel<AM_F32, EP_BIAS, DIN><<<dim3(M / 128, 2), 256, 0, s>>>(
      x, wbuf, M, 128, Tm, nullptr, nullptr, nullptr, P.b_in, nullptr, h, nullptr, nullptr, 128);
  for (int l = 0; l < 2; ++l) {
    const float* ln1g = P.ln1_g + l * 128;  const float* ln1b = P.ln1_b + l * 128;
    const float* ln2g = P.ln2_g + l * 128;  const float* ln2b = P.ln2_b + l * 128;
    const float* mk = P.mix_k + l * 128;    const float* mv = P.mix_v + l * 128;
    const float* mr = P.mix_r + l * 128;
    const float* cmk = P.cmix_k + l * 128;  const float* cmr = P.cmix_r + l * 128;
    const float* Wk = P.Wk + (size_t)l * 128 * 128;  const float* Wv = P.Wv + (size_t)l * 128 * 128;
    const float* Wr = P.Wr + (size_t)l * 128 * 128;  const float* Wo = P.Wo + (size_t)l * 128 * 128;
    const float* cWk = P.cWk + (size_t)l * 128 * 512;
    const float* cWr = P.cWr + (size_t)l * 128 * 128;
    const float* cWv = P.cWv + (size_t)l * 512 * 128;

    ln_kernel<<<M / 8, 256, 0, s>>>(h, ln1g, ln1b, z);
    pack3t_kernel<<<cdiv(384 * 128, 256), 256, 0, s>>>(Wk, Wv, Wr, wbuf);
    gemm_kernel<AM_F32MIX, EP_NONE, 128><<<dim3(M / 128, 6), 256, 0, s>>>(
        z, wbuf, M, 384, Tm, mk, mv, mr, nullptr, nullptr, kvr, nullptr, nullptr, 384);
    wkv_kernel<<<(16 * 128) / 256, 256, 0, s>>>(kvr, P.decay + l * 128, P.first + l * 128, ry, T);
    pack1t_kernel<<<cdiv(128 * 128, 256), 256, 0, s>>>(Wo, wbuf, 128, 128);
    gemm_kernel<AM_BF16, EP_ADD, 128><<<dim3(M / 128, 2), 256, 0, s>>>(
        ry, wbuf, M, 128, Tm, nullptr, nullptr, nullptr, nullptr, nullptr, nullptr, nullptr, h, 128);

    ln_kernel<<<M / 8, 256, 0, s>>>(h, ln2g, ln2b, z);
    pack1t_kernel<<<cdiv(128 * 512, 256), 256, 0, s>>>(cWk, wbuf, 128, 512);
    gemm_kernel<AM_F32MIX, EP_RELU2BF, 128><<<dim3(M / 128, 8), 256, 0, s>>>(
        z, wbuf, M, 512, Tm, cmk, cmk, cmk, nullptr, nullptr, nullptr, kk, nullptr, 512);
    pack1t_kernel<<<cdiv(128 * 128, 256), 256, 0, s>>>(cWr, wbuf, 128, 128);
    gemm_kernel<AM_F32MIX, EP_SIGMOID, 128><<<dim3(M / 128, 2), 256, 0, s>>>(
        z, wbuf, M, 128, Tm, cmr, cmr, cmr, nullptr, nullptr, rrb, nullptr, nullptr, 128);
    pack1t_kernel<<<cdiv(512 * 128, 256), 256, 0, s>>>(cWv, wbuf, 512, 128);
    gemm_kernel<AM_BF16, EP_MULADD, 512><<<dim3(M / 128, 2), 256, 0, s>>>(
        kk, wbuf, M, 128, Tm, nullptr, nullptr, nullptr, nullptr, rrb, nullptr, nullptr, h, 128);
  }
  lastln_kernel<<<1, 512, 0, s>>>(h, T, P.lnf_g, P.lnf_b, last_out);
}

extern "C" void kernel_launch(void* const* d_in, const int* in_sizes, int n_in,
                              void* d_out, int out_size, void* d_ws, size_t ws_size,
                              hipStream_t stream) {
  (void)in_sizes; (void)n_in; (void)out_size; (void)ws_size;
  int i = 0;
  const float* x = (const float*)d_in[i++];
  RwkvP fine = unpack_rwkv(d_in, i);
  RwkvP med  = unpack_rwkv(d_in, i);
  RwkvP coa  = unpack_rwkv(d_in, i);
  const float* projW = (const float*)d_in[i++];
  const float* projb = (const float*)d_in[i++];
  const float* Wq = (const float*)d_in[i++]; const float* bq = (const float*)d_in[i++];
  const float* Wk = (const float*)d_in[i++]; const float* bk = (const float*)d_in[i++];
  const float* Wv = (const float*)d_in[i++]; const float* bv = (const float*)d_in[i++];
  const float* Wo = (const float*)d_in[i++]; const float* bo = (const float*)d_in[i++];
  const float* ng = (const float*)d_in[i++]; const float* nb = (const float*)d_in[i++];
  (void)Wq; (void)bq; (void)Wk; (void)bk;   // softmax over length-1 axis == 1; q/k are dead

  char* ws = (char*)d_ws;
  size_t off = 0;
  auto carve = [&](size_t bytes) -> char* {
    char* p = ws + off; off += (bytes + 255) & ~(size_t)255; return p;
  };
  float*  pool_m = (float*) carve((size_t)16 * 2048 * 128 * 4);   // 16 MB
  float*  pool_c = (float*) carve((size_t)16 * 1024 * 64 * 4);    //  4 MB
  float*  h      = (float*) carve((size_t)65536 * 128 * 4);       // 32 MB (reused per branch)
  float*  z      = (float*) carve((size_t)65536 * 128 * 4);       // 32 MB
  float*  kvr    = (float*) carve((size_t)65536 * 384 * 4);       // 96 MB (kk/rr alias inside)
  bf16_t* ry     = (bf16_t*)carve((size_t)65536 * 128 * 2);       // 16 MB
  bf16_t* wbuf   = (bf16_t*)carve((size_t)512 * 512 * 2);         // packed-weight scratch
  float*  last   = (float*) carve((size_t)3 * 16 * 128 * 4);
  bf16_t* kk  = (bf16_t*)kvr;                                     // M*512 bf16 <= 64 MB
  float*  rrb = (float*)((char*)kvr + (size_t)65536 * 512 * 2);   // 32 MB tail of kvr region

  pool2_kernel<<<(16 * 2048 * 128) / 256, 256, 0, stream>>>(x, pool_m);
  pool4_kernel<<<(16 * 1024 * 64) / 256, 256, 0, stream>>>(x, pool_c);

  run_branch<256, 4096>(x,      fine, h, z, kvr, ry, kk, rrb, wbuf, last,                stream);
  run_branch<128, 2048>(pool_m, med,  h, z, kvr, ry, kk, rrb, wbuf, last + 16 * 128,     stream);
  run_branch< 64, 1024>(pool_c, coa,  h, z, kvr, ry, kk, rrb, wbuf, last + 2 * 16 * 128, stream);

  readout_kernel<<<16, 512, 0, stream>>>(last, projW, projb, Wv, bv, Wo, bo, ng, nb,
                                         (float*)d_out);
}
